// VariationalGraohAutoDecoder_2044404433055
// MI455X (gfx1250) — matrix operand order repaired and checked by reading, not compiled, not run on hardware
//
#include <hip/hip_runtime.h>

// ---------------------------------------------------------------------------
// GraphSAGE (3-layer SAGEConv + residuals) for MI455X / gfx1250, wave32.
//   phase 1: f32 atomic scatter-mean over edges (L2 resident)
//   phase 2: fused K-concat GEMM  [mean | x | z] @ [Wn; Wr; Wres] + b
//            via v_wmma_f32_16x16x32_bf16, MT=2 register blocking so each
//            LDS B-fragment feeds two WMMAs; weights staged to LDS with
//            global_load_async_to_lds_b128 (ASYNCcnt) when available.
// ---------------------------------------------------------------------------

typedef __attribute__((ext_vector_type(16))) __bf16 v16bf;
typedef __attribute__((ext_vector_type(8)))  __bf16 v8bf;
typedef __attribute__((ext_vector_type(8)))  float  v8f;
typedef __attribute__((ext_vector_type(4)))  int    v4i;

#define NN  50000
#define NE  800000
#define EMBD 128

union BFrag { v16bf v; v8bf h[2]; };

// --------------------------------------------- async LDS staging helpers ----
using as1_v4i = __attribute__((address_space(1))) v4i;  // global int4
using as3_v4i = __attribute__((address_space(3))) v4i;  // LDS int4

__device__ __forceinline__ void stage16(const void* g, void* l) {
#if __has_builtin(__builtin_amdgcn_global_load_async_to_lds_b128)
  // low 32 bits of a flat shared pointer are the LDS byte offset
  __builtin_amdgcn_global_load_async_to_lds_b128(
      (as1_v4i*)(unsigned long long)g,
      (as3_v4i*)(unsigned int)(unsigned long long)l,
      0, 0);
#else
  *(uint4*)l = *(const uint4*)g;
#endif
}

__device__ __forceinline__ void stage_wait() {
#if __has_builtin(__builtin_amdgcn_global_load_async_to_lds_b128)
#if __has_builtin(__builtin_amdgcn_s_wait_asynccnt)
  __builtin_amdgcn_s_wait_asynccnt(0);
#else
  asm volatile("s_wait_asynccnt 0x0" ::: "memory");
#endif
#endif
}

// ---------------------------------------------------------------- degree ----
__global__ void deg_count(const long long* __restrict__ dst, float* __restrict__ deg) {
  int e = blockIdx.x * blockDim.x + threadIdx.x;
  if (e < NE) atomicAdd(&deg[(int)dst[e]], 1.0f);
}

__global__ void inv_deg(float* __restrict__ deg) {
  int i = blockIdx.x * blockDim.x + threadIdx.x;
  if (i < NN) deg[i] = 1.0f / fmaxf(deg[i], 1.0f);
}

// --------------------------------------------------------------- scatter ----
// one thread = one edge x 4 channels; float4 gather + 4 f32 atomics
__global__ void scatter_add(const float* __restrict__ x,
                            const long long* __restrict__ src,
                            const long long* __restrict__ dst,
                            float* __restrict__ agg) {
  long long t = (long long)blockIdx.x * blockDim.x + threadIdx.x;
  if (t >= (long long)NE * 32) return;
  int e = (int)(t >> 5);
  int c = (int)(t & 31) * 4;
  int s = (int)src[e];
  int d = (int)dst[e];
  const float4 v = *(const float4*)(x + (size_t)s * EMBD + c);
  float* a = agg + (size_t)d * EMBD + c;
  atomicAdd(a + 0, v.x);
  atomicAdd(a + 1, v.y);
  atomicAdd(a + 2, v.z);
  atomicAdd(a + 3, v.w);
}

// --------------------------------------------------- A matrix assembly ------
__global__ void build_A(const float* __restrict__ agg, const float* __restrict__ invdeg,
                        const float* __restrict__ xcur, const float* __restrict__ z,
                        __bf16* __restrict__ A, int Ktot) {
  long long idx = (long long)blockIdx.x * blockDim.x + threadIdx.x;
  long long total = (long long)NN * Ktot;
  if (idx >= total) return;
  int k = (int)(idx % Ktot);
  int m = (int)(idx / Ktot);
  float v;
  if (k < EMBD)          v = agg[(size_t)m * EMBD + k] * invdeg[m];
  else if (k < 2 * EMBD) v = xcur[(size_t)m * EMBD + (k - EMBD)];
  else                   v = z[(size_t)m * EMBD + (k - 2 * EMBD)];
  A[idx] = (__bf16)v;
}

// ------------------------------------------------ weight pre-permutation ----
__device__ __forceinline__ void pack_one(int k, int n, int KSTEPS, float v,
                                         __bf16* __restrict__ dst) {
  int nt = n >> 4, n16 = n & 15;
  int kk = k >> 5, kr = k & 31;
  int lane = ((kr >> 4) << 4) + n16;
  int t = kr & 15;
  dst[(((nt * KSTEPS + kk) * 32 + lane) << 4) + t] = (__bf16)v;
}

__global__ void pack_weights(const float* __restrict__ w1n, const float* __restrict__ w1r,
                             const float* __restrict__ w2n, const float* __restrict__ w2r,
                             const float* __restrict__ wr1,
                             const float* __restrict__ wfn, const float* __restrict__ wfr,
                             const float* __restrict__ wr2,
                             __bf16* __restrict__ W1p, __bf16* __restrict__ W2p,
                             __bf16* __restrict__ WFp) {
  int idx = blockIdx.x * blockDim.x + threadIdx.x;
  if (idx < 256 * 128) {                       // layer 1: K=256, N=128
    int k = idx >> 7, n = idx & 127;
    float v = (k < 128) ? w1n[k * 128 + n] : w1r[(k - 128) * 128 + n];
    pack_one(k, n, 8, v, W1p);
  } else if (idx < 256 * 128 + 384 * 128) {    // layer 2: K=384, N=128
    int i = idx - 256 * 128;
    int k = i >> 7, n = i & 127;
    float v = (k < 128) ? w2n[k * 128 + n]
            : (k < 256) ? w2r[(k - 128) * 128 + n]
                        : wr1[(k - 256) * 128 + n];
    pack_one(k, n, 12, v, W2p);
  } else if (idx < 256 * 128 + 384 * 128 + 384 * 64) {  // final: K=384, N=64
    int i = idx - (256 * 128 + 384 * 128);
    int k = i / 64, n = i % 64;
    float v = (k < 128) ? wfn[k * 64 + n]
            : (k < 256) ? wfr[(k - 128) * 64 + n]
                        : wr2[(k - 256) * 64 + n];
    pack_one(k, n, 12, v, WFp);
  }
}

__global__ void make_bias(const float* __restrict__ b1, const float* __restrict__ b2,
                          const float* __restrict__ br1, const float* __restrict__ bfv,
                          const float* __restrict__ br2, float* __restrict__ bias) {
  int i = threadIdx.x;
  if (i < 128)      bias[i] = b1[i];
  else if (i < 256) bias[i] = b2[i - 128] + br1[i - 128];
  else if (i < 320) bias[i] = bfv[i - 256] + br2[i - 256];
}

// ----------------------------------------------------------- WMMA GEMM ------
// One wave computes TWO 16-row x N tiles (MT=2): each B fragment read from
// LDS feeds two independent back-to-back WMMAs.
template <int NT, int KSTEPS>
__global__ void __launch_bounds__(256)
sage_gemm(const __bf16* __restrict__ A, const __bf16* __restrict__ Bp,
          const float* __restrict__ bias, float* __restrict__ out,
          int M, int relu, int sig8) {
  constexpr int Ktot = KSTEPS * 32;
  constexpr int N = NT * 16;
  extern __shared__ char smem_raw[];

  // cooperative stage of packed weights into LDS (NT*KSTEPS*1024 bytes)
  for (int i = threadIdx.x; i < NT * KSTEPS * 64; i += 256)
    stage16((const char*)Bp + i * 16, smem_raw + i * 16);
  stage_wait();
  __syncthreads();

  const int wave = threadIdx.x >> 5;
  const int lane = threadIdx.x & 31;
  const int lane16 = lane & 15;
  const int laneHi = lane >> 4;

  const int pair = blockIdx.x * 8 + wave;   // pair of 16-row tiles
  const int rowBase0 = pair * 32;
  const int rowBase1 = rowBase0 + 16;
  if (rowBase0 >= M) return;  // uniform per wave; EXEC stays all-ones

  const int r0 = rowBase0 + lane16;                       // always < M
  const int r1 = min(rowBase1 + lane16, M - 1);           // clamp for tail
  const __bf16* Arow0 = A + (size_t)r0 * Ktot;
  const __bf16* Arow1 = A + (size_t)r1 * Ktot;
  const __bf16* sB = (const __bf16*)smem_raw;

  v8f zero = {};
  v8f acc0[NT], acc1[NT];
#pragma unroll
  for (int nt = 0; nt < NT; ++nt) { acc0[nt] = zero; acc1[nt] = zero; }

#pragma unroll
  for (int kk = 0; kk < KSTEPS; ++kk) {
    BFrag a0, a1;
    const __bf16* ap0 = Arow0 + kk * 32 + laneHi * 16;
    const __bf16* ap1 = Arow1 + kk * 32 + laneHi * 16;
    a0.h[0] = *(const v8bf*)(ap0);        // K runs {0..7} / {16..23}
    a0.h[1] = *(const v8bf*)(ap0 + 8);    // K runs {8..15} / {24..31}
    a1.h[0] = *(const v8bf*)(ap1);
    a1.h[1] = *(const v8bf*)(ap1 + 8);
    if (kk + 1 < KSTEPS) {
      __builtin_prefetch(Arow0 + (kk + 1) * 32, 0, 3);
      __builtin_prefetch(Arow1 + (kk + 1) * 32, 0, 3);
    }
#pragma unroll
    for (int nt = 0; nt < NT; ++nt) {
      BFrag b;
      const __bf16* bp = sB + (((nt * KSTEPS + kk) * 32 + lane) << 4);
      b.h[0] = *(const v8bf*)(bp);
      b.h[1] = *(const v8bf*)(bp + 8);
      acc0[nt] = __builtin_amdgcn_wmma_f32_16x16x32_bf16(
          false, a0.v, false, b.v, (short)0, acc0[nt], false, false);
      acc1[nt] = __builtin_amdgcn_wmma_f32_16x16x32_bf16(
          false, a1.v, false, b.v, (short)0, acc1[nt], false, false);
    }
  }

  // epilogue: bias (+relu) (+sigmoid on cols 0..7), f32 stores
#pragma unroll
  for (int nt = 0; nt < NT; ++nt) {
    int col = nt * 16 + lane16;
    float bb = bias[col];
#pragma unroll
    for (int r = 0; r < 8; ++r) {
      float v = acc0[nt][r] + bb;
      if (relu) v = fmaxf(v, 0.0f);
      if (sig8 && col < 8) v = 1.0f / (1.0f + __expf(-v));
      out[(size_t)(rowBase0 + laneHi * 8 + r) * N + col] = v;
    }
  }
  if (rowBase1 < M) {
#pragma unroll
    for (int nt = 0; nt < NT; ++nt) {
      int col = nt * 16 + lane16;
      float bb = bias[col];
#pragma unroll
      for (int r = 0; r < 8; ++r) {
        float v = acc1[nt][r] + bb;
        if (relu) v = fmaxf(v, 0.0f);
        if (sig8 && col < 8) v = 1.0f / (1.0f + __expf(-v));
        out[(size_t)(rowBase1 + laneHi * 8 + r) * N + col] = v;
      }
    }
  }
}

// ------------------------------------------------------------- launcher -----
extern "C" void kernel_launch(void* const* d_in, const int* in_sizes, int n_in,
                              void* d_out, int out_size, void* d_ws, size_t ws_size,
                              hipStream_t stream) {
  const float* z       = (const float*)d_in[0];
  const long long* ei  = (const long long*)d_in[1];  // int64 per reference
  const float* w1n = (const float*)d_in[2];
  const float* w1r = (const float*)d_in[3];
  const float* b1  = (const float*)d_in[4];
  const float* w2n = (const float*)d_in[5];
  const float* w2r = (const float*)d_in[6];
  const float* b2  = (const float*)d_in[7];
  const float* wfn = (const float*)d_in[8];
  const float* wfr = (const float*)d_in[9];
  const float* bfv = (const float*)d_in[10];
  const float* wr1 = (const float*)d_in[11];
  const float* br1 = (const float*)d_in[12];
  const float* wr2 = (const float*)d_in[13];
  const float* br2 = (const float*)d_in[14];

  const long long* src = ei;
  const long long* dst = ei + NE;

  char* ws = (char*)d_ws;
  size_t off = 0;
  auto alloc = [&](size_t bytes) -> char* {
    char* p = ws + off;
    off += (bytes + 255) & ~(size_t)255;
    return p;
  };
  float*  deg  = (float*)alloc((size_t)NN * 4);
  float*  agg  = (float*)alloc((size_t)NN * EMBD * 4);
  float*  h1   = (float*)alloc((size_t)NN * EMBD * 4);
  float*  h2   = (float*)alloc((size_t)NN * EMBD * 4);
  __bf16* Abuf = (__bf16*)alloc((size_t)NN * 384 * 2);
  __bf16* W1p  = (__bf16*)alloc(256 * 128 * 2);
  __bf16* W2p  = (__bf16*)alloc(384 * 128 * 2);
  __bf16* WFp  = (__bf16*)alloc(384 * 64 * 2);
  float*  bias = (float*)alloc(320 * 4);

  // degree / inverse degree (graph-constant across layers)
  (void)hipMemsetAsync(deg, 0, (size_t)NN * 4, stream);
  deg_count<<<(NE + 255) / 256, 256, 0, stream>>>(dst, deg);
  inv_deg<<<(NN + 255) / 256, 256, 0, stream>>>(deg);

  // weight permutation + fused biases
  pack_weights<<<(256 * 128 + 384 * 128 + 384 * 64 + 255) / 256, 256, 0, stream>>>(
      w1n, w1r, w2n, w2r, wr1, wfn, wfr, wr2, W1p, W2p, WFp);
  make_bias<<<1, 320, 0, stream>>>(b1, b2, br1, bfv, br2, bias);

  // 3125 row tiles -> 1563 tile pairs -> 8 pairs/block
  const int gemmGrid = ((NN / 16 + 1) / 2 + 7) / 8;   // 196
  const long long sthreads = (long long)NE * 32;
  const int sgrid = (int)((sthreads + 255) / 256);

  // ---- layer 1: h1 = relu([mean(z)|z] @ [w1n;w1r] + b1) ----
  (void)hipMemsetAsync(agg, 0, (size_t)NN * EMBD * 4, stream);
  scatter_add<<<sgrid, 256, 0, stream>>>(z, src, dst, agg);
  build_A<<<(int)(((long long)NN * 256 + 255) / 256), 256, 0, stream>>>(
      agg, deg, z, z, Abuf, 256);
  sage_gemm<8, 8><<<gemmGrid, 256, 8 * 8 * 1024, stream>>>(
      Abuf, W1p, bias, h1, NN, 1, 0);

  // ---- layer 2: h2 = relu([mean(h1)|h1|z] @ [w2n;w2r;wres1] + b2+bres1) ----
  (void)hipMemsetAsync(agg, 0, (size_t)NN * EMBD * 4, stream);
  scatter_add<<<sgrid, 256, 0, stream>>>(h1, src, dst, agg);
  build_A<<<(int)(((long long)NN * 384 + 255) / 256), 256, 0, stream>>>(
      agg, deg, h1, z, Abuf, 384);
  sage_gemm<8, 12><<<gemmGrid, 256, 8 * 12 * 1024, stream>>>(
      Abuf, W2p, bias + 128, h2, NN, 1, 0);

  // ---- final: out = [mean(h2)|h2|z] @ [wfn;wfr;wres2] + bf+bres2, sigmoid[:8]
  (void)hipMemsetAsync(agg, 0, (size_t)NN * EMBD * 4, stream);
  scatter_add<<<sgrid, 256, 0, stream>>>(h2, src, dst, agg);
  build_A<<<(int)(((long long)NN * 384 + 255) / 256), 256, 0, stream>>>(
      agg, deg, h2, z, Abuf, 384);
  sage_gemm<4, 12><<<gemmGrid, 256, 4 * 12 * 1024, stream>>>(
      Abuf, WFp, bias + 256, (float*)d_out, NN, 0, 1);
}
